// HolE_62878321214080
// MI455X (gfx1250) — matrix operand order repaired and checked by reading, not compiled, not run on hardware
//
#include <hip/hip_runtime.h>
#include <math.h>

typedef __attribute__((ext_vector_type(16))) _Float16 v16h;
typedef __attribute__((ext_vector_type(8)))  _Float16 h8;
typedef __attribute__((ext_vector_type(4)))  _Float16 h4;
typedef __attribute__((ext_vector_type(8)))  float    v8f;

#define HIDDEN     128
#define BATCH      32768
#define NNEG       8
#define TOTAL      (BATCH * (1 + NNEG))
#define NFREQ      80            // 65 real frequencies padded to 5 tiles of 16
#define SROW       136           // LDS row stride in halves (128 + 8 pad, bank-conflict free)
#define TRI_PER_WG 128
#define SCORE_OFF  65536         // scores start here in d_ws (W table is 43,520 B)

// ---------------------------------------------------------------------------
// Kernel 0: build padded DFT table W (f16) in global scratch.
// Rows 0..79   : Wre[n][k] =  cos(2*pi*n*k/128)
// Rows 80..159 : Wim[n][k] = -sin(2*pi*n*k/128)
// Columns 128..135 are zero padding. Rows n>=65 get weight 0 at combine time.
// Phase reduced exactly via (n*k) mod 128.
// ---------------------------------------------------------------------------
__global__ void build_dft(_Float16* __restrict__ W) {
    int idx = blockIdx.x * 256 + threadIdx.x;
    const int total = 2 * NFREQ * SROW;
    if (idx >= total) return;
    int row = idx / SROW;
    int k   = idx % SROW;
    float v = 0.0f;
    if (k < HIDDEN) {
        int n  = (row < NFREQ) ? row : row - NFREQ;
        int ph = (n * k) & (HIDDEN - 1);
        float th = (float)ph * (6.2831853071795864769f / 128.0f);
        v = (row < NFREQ) ? __cosf(th) : -__sinf(th);
    }
    W[idx] = (_Float16)v;
}

// ---------------------------------------------------------------------------
// Kernel 1: fused gather + DFT-GEMM (WMMA f16 16x16x32) + spectral combine.
// One workgroup = 256 threads = 8 waves, 128 triples. Each wave owns 16
// triples and produces their scalar scores.
// ---------------------------------------------------------------------------
__global__ __launch_bounds__(256) void hole_score(
    const int*   __restrict__ bh,
    const int*   __restrict__ bt,
    const int*   __restrict__ br,
    const float* __restrict__ ent,
    const float* __restrict__ rel,
    const _Float16* __restrict__ gW,
    float* __restrict__ scores)
{
    extern __shared__ __align__(16) char smem_raw[];
    _Float16* sW = (_Float16*)smem_raw;                    // 160 * 136 halves
    _Float16* sE = sW + 2 * NFREQ * SROW;                  // 384 * 136 halves
    float* sScale = (float*)(sE + 3 * TRI_PER_WG * SROW);  // 128 floats

    const int tid  = threadIdx.x;
    const int base = blockIdx.x * TRI_PER_WG;

    // --- Stage A: copy DFT table into LDS (43,520 B = 2720 x uint4) ---
    {
        const uint4* src = (const uint4*)gW;
        uint4*       dst = (uint4*)sW;
        for (int i = tid; i < (2 * NFREQ * SROW) / 8; i += 256) dst[i] = src[i];
    }

    // --- Stage B: gather h/t/r rows, convert f32 -> f16 into padded LDS ---
    // rows 0..127 = head, 128..255 = tail, 256..383 = rel
    for (int e = tid; e < 3 * TRI_PER_WG * (HIDDEN / 4); e += 256) {
        int row = e >> 5;          // 32 float4 per row
        int c4  = e & 31;
        int s   = row >> 7;
        int lt  = row & 127;
        int gidx;
        const float* tab;
        if (s == 0)      { gidx = bh[base + lt]; tab = ent; }
        else if (s == 1) { gidx = bt[base + lt]; tab = ent; }
        else             { gidx = br[base + lt]; tab = rel; }
        float4 v = ((const float4*)(tab + (size_t)gidx * HIDDEN))[c4];
        h4 hv = { (_Float16)v.x, (_Float16)v.y, (_Float16)v.z, (_Float16)v.w };
        *(h4*)(sE + row * SROW + c4 * 4) = hv;
    }
    __syncthreads();

    // --- Stage B2: per-triple rel norm scale (score is linear in rel) ---
    if (tid < TRI_PER_WG) {
        const _Float16* rrow = sE + (2 * TRI_PER_WG + tid) * SROW;
        float ss = 0.0f;
        for (int k = 0; k < HIDDEN; ++k) { float x = (float)rrow[k]; ss += x * x; }
        sScale[tid] = rsqrtf(fmaxf(ss, 1e-12f));
    }
    __syncthreads();

    // --- Stage C: WMMA DFT + frequency-domain combine ---
    const int wave = tid >> 5;
    const int lane = tid & 31;
    const int m    = lane & 15;   // row / column within 16
    const int hi   = lane >> 4;   // lane half

    const _Float16* rowA = sE + (0 * TRI_PER_WG + wave * 16 + m) * SROW;
    const _Float16* rowB = sE + (1 * TRI_PER_WG + wave * 16 + m) * SROW;
    const _Float16* rowR = sE + (2 * TRI_PER_WG + wave * 16 + m) * SROW;

    float sum8[8];
#pragma unroll
    for (int v = 0; v < 8; ++v) sum8[v] = 0.0f;

    for (int nt = 0; nt < 5; ++nt) {
        const _Float16* wre = sW + (nt * 16 + m) * SROW;
        const _Float16* wim = sW + (NFREQ + nt * 16 + m) * SROW;

        v8f aR = {}, aI = {}, bR = {}, bI = {}, rR = {}, rI = {};
#pragma unroll
        for (int kb = 0; kb < 4; ++kb) {
            const int ks = kb * 32;
            // A-operand layout: lane half 0 -> K {ks..ks+7, ks+16..ks+23}
            //                   lane half 1 -> K {ks+8..ks+15, ks+24..ks+31}
            const int oa0 = ks + 8 * hi, oa1 = ks + 16 + 8 * hi;
            // B-operand layout: same as A (half 0 -> K {ks..ks+7, ks+16..ks+23}, half 1 -> K {ks+8..ks+15, ks+24..ks+31})
            const int ob0 = ks + 8 * hi, ob1 = ks + 16 + 8 * hi;

            h8 a0 = *(const h8*)(rowA + oa0), a1 = *(const h8*)(rowA + oa1);
            h8 b0 = *(const h8*)(rowB + oa0), b1 = *(const h8*)(rowB + oa1);
            h8 r0 = *(const h8*)(rowR + oa0), r1 = *(const h8*)(rowR + oa1);
            h8 c0 = *(const h8*)(wre + ob0),  c1 = *(const h8*)(wre + ob1);
            h8 s0 = *(const h8*)(wim + ob0),  s1 = *(const h8*)(wim + ob1);

            v16h A  = __builtin_shufflevector(a0, a1, 0,1,2,3,4,5,6,7,8,9,10,11,12,13,14,15);
            v16h B  = __builtin_shufflevector(b0, b1, 0,1,2,3,4,5,6,7,8,9,10,11,12,13,14,15);
            v16h R  = __builtin_shufflevector(r0, r1, 0,1,2,3,4,5,6,7,8,9,10,11,12,13,14,15);
            v16h Wc = __builtin_shufflevector(c0, c1, 0,1,2,3,4,5,6,7,8,9,10,11,12,13,14,15);
            v16h Ws = __builtin_shufflevector(s0, s1, 0,1,2,3,4,5,6,7,8,9,10,11,12,13,14,15);

            aR = __builtin_amdgcn_wmma_f32_16x16x32_f16(false, A, false, Wc, (short)0, aR, false, false);
            aI = __builtin_amdgcn_wmma_f32_16x16x32_f16(false, A, false, Ws, (short)0, aI, false, false);
            bR = __builtin_amdgcn_wmma_f32_16x16x32_f16(false, B, false, Wc, (short)0, bR, false, false);
            bI = __builtin_amdgcn_wmma_f32_16x16x32_f16(false, B, false, Ws, (short)0, bI, false, false);
            rR = __builtin_amdgcn_wmma_f32_16x16x32_f16(false, R, false, Wc, (short)0, rR, false, false);
            rI = __builtin_amdgcn_wmma_f32_16x16x32_f16(false, R, false, Ws, (short)0, rI, false, false);
        }

        // frequency weight: f=0,64 -> 1 ; f=1..63 -> 2 ; padding f>=65 -> 0
        const int n = nt * 16 + m;
        const float wf = (n == 0 || n == 64) ? 1.0f : (n < 64 ? 2.0f : 0.0f);

        // Re[conj(FA)*FB*conj(FR)] = pr*rr + pi*ri, with
        // pr = ar*br + ai*bi ; pi = ar*bi - ai*br
        // (identical (row,col)->(lane,vgpr) layout across all 6 accumulators)
#pragma unroll
        for (int v = 0; v < 8; ++v) {
            float pr = aR[v] * bR[v] + aI[v] * bI[v];
            float pi = aR[v] * bI[v] - aI[v] * bR[v];
            sum8[v] += wf * (pr * rR[v] + pi * rI[v]);
        }
    }

    // Reduce over 16 columns (lanes within each half-wave).
#pragma unroll
    for (int v = 0; v < 8; ++v) {
        sum8[v] += __shfl_xor(sum8[v], 1, 32);
        sum8[v] += __shfl_xor(sum8[v], 2, 32);
        sum8[v] += __shfl_xor(sum8[v], 4, 32);
        sum8[v] += __shfl_xor(sum8[v], 8, 32);
    }

    // lanes 0..15 now all hold rows 0..7 ; lanes 16..31 hold rows 8..15.
    if (m == 0) {
#pragma unroll
        for (int v = 0; v < 8; ++v) {
            int lt = wave * 16 + hi * 8 + v;
            float inner = sum8[v] * (1.0f / 128.0f) * sScale[lt];
            scores[base + lt] = -1.0f / (1.0f + __expf(-inner));
        }
    }
}

// ---------------------------------------------------------------------------
// Kernel 2: hinge loss, single block (deterministic reduction).
// ---------------------------------------------------------------------------
__global__ __launch_bounds__(256) void hole_loss(const float* __restrict__ scores,
                                                 float* __restrict__ out)
{
    __shared__ float red[256];
    float acc = 0.0f;
    for (int i = threadIdx.x; i < BATCH; i += 256) {
        float p = scores[i];
        const float* nb = scores + BATCH + i * NNEG;
        float ns = 0.0f;
#pragma unroll
        for (int j = 0; j < NNEG; ++j) ns += nb[j];
        ns *= (1.0f / NNEG);
        acc += fmaxf(p - ns + 1.0f, 0.0f);
    }
    red[threadIdx.x] = acc;
    __syncthreads();
    for (int s = 128; s > 0; s >>= 1) {
        if (threadIdx.x < s) red[threadIdx.x] += red[threadIdx.x + s];
        __syncthreads();
    }
    if (threadIdx.x == 0) out[0] = red[0];
}

// ---------------------------------------------------------------------------
extern "C" void kernel_launch(void* const* d_in, const int* in_sizes, int n_in,
                              void* d_out, int out_size, void* d_ws, size_t ws_size,
                              hipStream_t stream) {
    (void)in_sizes; (void)n_in; (void)out_size; (void)ws_size;

    const int*   bh  = (const int*)d_in[0];
    const int*   bt  = (const int*)d_in[1];
    const int*   br  = (const int*)d_in[2];
    const float* ent = (const float*)d_in[3];
    const float* rel = (const float*)d_in[4];

    _Float16* gW     = (_Float16*)d_ws;
    float*    scores = (float*)((char*)d_ws + SCORE_OFF);
    float*    out    = (float*)d_out;

    const int wtot = 2 * NFREQ * SROW;
    build_dft<<<(wtot + 255) / 256, 256, 0, stream>>>(gW);

    const size_t lds = (size_t)(2 * NFREQ + 3 * TRI_PER_WG) * SROW * sizeof(_Float16)
                     + TRI_PER_WG * sizeof(float);   // ~148 KB (CDNA5: up to 320 KB/WG)
    hole_score<<<TOTAL / TRI_PER_WG, 256, lds, stream>>>(bh, bt, br, ent, rel, gW, scores);

    hole_loss<<<1, 256, 0, stream>>>(scores, out);
}
